// LearnedUpsampling_63307817943571
// MI455X (gfx1250) — matrix operand order repaired and checked by reading, not compiled, not run on hardware
//
#include <hip/hip_runtime.h>

#define B_  8
#define K_  256
#define D_  256
#define T_  1000
#define NTILE 63   // ceil(1000/16)

typedef __attribute__((ext_vector_type(16))) _Float16 v16h;
typedef __attribute__((ext_vector_type(8)))  _Float16 v8h;
typedef __attribute__((ext_vector_type(8)))  float    v8f;
typedef __attribute__((ext_vector_type(4)))  float    v4f;

union H16 { v16h v; _Float16 e[16]; };
union H8  { v8h  v; _Float16 e[8];  };
union F8  { v8f  v; float    e[8];  };
union F4  { v4f  v; float    e[4];  };

__device__ __forceinline__ float fast_rcp(float x) { return __builtin_amdgcn_rcpf(x); }
__device__ __forceinline__ float fast_rsq(float x) { return __builtin_amdgcn_rsqf(x); }
__device__ __forceinline__ float silu_f(float x) {
    return x * __builtin_amdgcn_rcpf(1.0f + __expf(-x));
}

// A-matrix (16x32 f16) per-lane K index for element i
__device__ __forceinline__ int a_kidx(int i, bool hi) {
    return i + (hi ? 16 : 0);
}

// ---------------------------------------------------------------------------
// Kernel 1: per-batch duration scan -> s_k, e_k, mel_len, mel_mask outputs
// ---------------------------------------------------------------------------
__global__ void __launch_bounds__(256)
k_prep(const float* __restrict__ dur,
       float* __restrict__ sk, float* __restrict__ ek,
       int* __restrict__ mlen_ws,
       float* __restrict__ out_mask, float* __restrict__ out_mlen)
{
    __shared__ float sc[256];
    __shared__ int mlS;
    const int b = blockIdx.x, k = threadIdx.x;
    const float v = dur[b * K_ + k];
    sc[k] = v;
    __syncthreads();
    for (int off = 1; off < 256; off <<= 1) {
        float t = (k >= off) ? sc[k - off] : 0.0f;
        __syncthreads();
        sc[k] += t;
        __syncthreads();
    }
    const float e = sc[k];
    ek[b * K_ + k] = e;
    sk[b * K_ + k] = e - v;
    if (k == 255) {
        int ml = (int)rintf(e);           // round-half-even like jnp.round
        ml = min(max(ml, 0), T_);
        mlS = ml;
        mlen_ws[b] = ml;
        out_mlen[b] = (float)ml;
    }
    __syncthreads();
    const int ml = mlS;
    for (int t = k; t < T_; t += 256)
        out_mask[b * T_ + t] = (t >= ml) ? 1.0f : 0.0f;
}

// ---------------------------------------------------------------------------
// Kernel 2: ConvBlock for both paths: conv1d(3, same) + bias -> SiLU -> LN(8)
// ---------------------------------------------------------------------------
__global__ void __launch_bounds__(256)
k_conv(const float* __restrict__ V,
       const float* __restrict__ kw_g, const float* __restrict__ bw_g,
       const float* __restrict__ sw_g, const float* __restrict__ sbw_g,
       const float* __restrict__ kc_g, const float* __restrict__ bc_g,
       const float* __restrict__ sc_g, const float* __restrict__ sbc_g,
       float* __restrict__ Vw, float* __restrict__ Vc)
{
    __shared__ float kwL[3 * 256 * 8];
    __shared__ float kcL[3 * 256 * 8];
    const int b = blockIdx.x, tid = threadIdx.x;
    for (int i = tid; i < 6144; i += 256) { kwL[i] = kw_g[i]; kcL[i] = kc_g[i]; }
    __syncthreads();

    const int k = tid;
    float aw[8], ac[8];
#pragma unroll
    for (int c = 0; c < 8; ++c) { aw[c] = bw_g[c]; ac[c] = bc_g[c]; }

    for (int tap = 0; tap < 3; ++tap) {
        const int kr = k + tap - 1;
        if (kr < 0 || kr >= K_) continue;
        const float* vr = V + ((size_t)b * K_ + kr) * D_;
        for (int cin = 0; cin < D_; cin += 4) {
            F4 xv; xv.v = *(const v4f*)(vr + cin);
#pragma unroll
            for (int q = 0; q < 4; ++q) {
                const float x = xv.e[q];
                const float* ww = &kwL[(tap * 256 + cin + q) * 8];
                const float* wc = &kcL[(tap * 256 + cin + q) * 8];
#pragma unroll
                for (int c = 0; c < 8; ++c) {
                    aw[c] = fmaf(x, ww[c], aw[c]);
                    ac[c] = fmaf(x, wc[c], ac[c]);
                }
            }
        }
    }
    // SiLU then LayerNorm(8)
    float muw = 0.f, muc = 0.f;
#pragma unroll
    for (int c = 0; c < 8; ++c) {
        aw[c] = silu_f(aw[c]); ac[c] = silu_f(ac[c]);
        muw += aw[c]; muc += ac[c];
    }
    muw *= 0.125f; muc *= 0.125f;
    float vw = 0.f, vc = 0.f;
#pragma unroll
    for (int c = 0; c < 8; ++c) {
        float dw = aw[c] - muw, dc = ac[c] - muc;
        vw += dw * dw; vc += dc * dc;
    }
    const float iw = fast_rsq(vw * 0.125f + 1e-5f);
    const float ic = fast_rsq(vc * 0.125f + 1e-5f);
    float* ow = Vw + ((size_t)b * K_ + k) * 8;
    float* oc = Vc + ((size_t)b * K_ + k) * 8;
#pragma unroll
    for (int c = 0; c < 8; ++c) {
        ow[c] = (aw[c] - muw) * iw * sw_g[c] + sbw_g[c];
        oc[c] = (ac[c] - muc) * ic * sc_g[c] + sbc_g[c];
    }
}

// ---------------------------------------------------------------------------
// Kernel 3: transpose V [b][k][d] f32 -> Vt [b][d][k] f16 (WMMA B operand)
// ---------------------------------------------------------------------------
__global__ void __launch_bounds__(256)
k_tr(const float* __restrict__ V, _Float16* __restrict__ Vt)
{
    const int idx = blockIdx.x * 256 + threadIdx.x;   // ((b*256 + d)*256 + k)
    const int k = idx & 255;
    const int d = (idx >> 8) & 255;
    const int b = idx >> 16;
    Vt[idx] = (_Float16)V[(((size_t)b * K_) + k) * D_ + d];
}

// ---------------------------------------------------------------------------
// Kernel 4: fused main kernel — one block per (t-tile of 16, batch)
// ---------------------------------------------------------------------------
__global__ void __launch_bounds__(256)
k_main(const float* __restrict__ sk_g, const float* __restrict__ ek_g,
       const int* __restrict__ mlen_g, const int* __restrict__ slen_g,
       const float* __restrict__ Vw_g, const float* __restrict__ Vc_g,
       const _Float16* __restrict__ Vt_g,
       const float* __restrict__ w1_g, const float* __restrict__ b1_g,
       const float* __restrict__ w2_g, const float* __restrict__ b2_g,
       const float* __restrict__ lw_g, const float* __restrict__ lb_g,
       const float* __restrict__ wc1_g, const float* __restrict__ bc1_g,
       const float* __restrict__ wc2_g, const float* __restrict__ bc2_g,
       const float* __restrict__ le_g, const float* __restrict__ leb_g,
       const float* __restrict__ lns_g, const float* __restrict__ lnb_g,
       float* __restrict__ out_up, float* __restrict__ out_W)
{
    __shared__ float skL[256], ekL[256];
    __shared__ int   maskL[256];
    __shared__ float bufA[4096];     // phases 1-4: Vw[0..2047], Vc[2048..4095]; 5-6: up[16][256]
    __shared__ float scoreL[4096];   // scores then W (f32) [t][k]
    __shared__ _Float16 WH[4096];    // W as f16 [t][k]
    __shared__ float cpart[16 * 16 * 2];
    __shared__ float ctx[32];        // [t][2]
    __shared__ float cw1[160], cb1[16], cw2[256], cb2[16], clw[16];
    __shared__ float cwc1[20], cbc1[2], cwc2[4], cbc2[2];

    const int b   = blockIdx.y;
    const int t0  = blockIdx.x * 16;
    const int tid = threadIdx.x;
    const int lane = tid & 31;
    const int w    = tid >> 5;
    const bool hi  = lane >= 16;
    const int lr   = lane & 15;

    const int mel_len = mlen_g[b];
    const int slen    = slen_g[b];
    const float lb0   = lb_g[0];

    // ---- stage per-k data + tiny weights into LDS ----
    {
        int i = tid;
        skL[i] = sk_g[b * K_ + i];
        ekL[i] = ek_g[b * K_ + i];
        maskL[i] = (i >= slen) ? 1 : 0;
        for (int j = tid; j < 2048; j += 256) {
            bufA[j]        = Vw_g[(size_t)b * 2048 + j];
            bufA[2048 + j] = Vc_g[(size_t)b * 2048 + j];
        }
        if (tid < 160) cw1[tid] = w1_g[tid];
        if (tid < 16)  cb1[tid] = b1_g[tid];
        if (tid < 256) cw2[tid] = w2_g[tid];
        if (tid < 16)  cb2[tid] = b2_g[tid];
        if (tid < 16)  clw[tid] = lw_g[tid];
        if (tid < 20)  cwc1[tid] = wc1_g[tid];
        if (tid < 2)   cbc1[tid] = bc1_g[tid];
        if (tid < 4)   cwc2[tid] = wc2_g[tid];
        if (tid < 2)   cbc2[tid] = bc2_g[tid];
    }
    __syncthreads();

    // ---- phase 2: scores via WMMA (transposed MLP, split-K channel trick),
    //      two t-rows interleaved so independent WMMA chains cover the
    //      f16-WMMA->VALU hazard slots with useful work instead of v_nops.
    H16 A1, A2;
#pragma unroll
    for (int i = 0; i < 16; ++i) {
        const int Ka = a_kidx(i, hi);
        float a1v = 0.0f, a2v = 0.0f;
        if (Ka < 5)                    a1v = cw1[Ka * 16 + lr];
        else if (Ka >= 16 && Ka < 21)  a1v = cw1[(Ka - 11) * 16 + lr];
        if (Ka < 8)                    a2v = cw2[Ka * 16 + lr];
        else if (Ka >= 16 && Ka < 24)  a2v = cw2[(Ka - 8) * 16 + lr];
        A1.e[i] = (_Float16)a1v;
        A2.e[i] = (_Float16)a2v;
    }
    // hoisted per-lane biases / weights (rows = channels (hi?8:0)+j)
    float bb1[8], bb2[8], lwv[8];
#pragma unroll
    for (int j = 0; j < 8; ++j) {
        const int ch = (hi ? 8 : 0) + j;
        bb1[j] = cb1[ch];
        bb2[j] = cb2[ch];
        lwv[j] = clw[ch];
    }
    const v8f zacc = {};

    const int tlA = w * 2;                 // this wave's two t-rows
    const int ta  = t0 + tlA;
    const int tb  = ta + 1;
    const bool deadA = (ta >= mel_len) || (ta >= T_);
    const bool deadB = (tb >= mel_len) || (tb >= T_);
    const float tva = (float)(ta + 1), tvb = (float)(tb + 1);

    for (int kc = 0; kc < 16; ++kc) {
        const int k = kc * 16 + lr;        // this lane's position (column)
        const int msk = maskL[k];
        const float skk = skL[k], ekk = ekL[k];
        F4 vw0, vw1;
        vw0.v = *(const v4f*)&bufA[k * 8];
        vw1.v = *(const v4f*)&bufA[k * 8 + 4];

        const bool ma = (msk != 0) || deadA;
        const bool mb = (msk != 0) || deadB;
        const float Sa = ma ? 0.0f : tva - skk;
        const float Ea = ma ? 0.0f : ekk - tva;
        const float Sb = mb ? 0.0f : tvb - skk;
        const float Eb = mb ? 0.0f : ekk - tvb;

        // lower half supplies features 0..4, upper half features 5..9
        H16 B1a, B1b;
        {
            const float c2v = hi ? vw1.e[1] : vw0.e[0];
            const float c3v = hi ? vw1.e[2] : vw0.e[1];
            const float c4v = hi ? vw1.e[3] : vw0.e[2];
            B1a.e[0] = (_Float16)(hi ? vw0.e[3] : Sa);
            B1a.e[1] = (_Float16)(hi ? vw1.e[0] : Ea);
            B1b.e[0] = (_Float16)(hi ? vw0.e[3] : Sb);
            B1b.e[1] = (_Float16)(hi ? vw1.e[0] : Eb);
            B1a.e[2] = B1b.e[2] = (_Float16)c2v;
            B1a.e[3] = B1b.e[3] = (_Float16)c3v;
            B1a.e[4] = B1b.e[4] = (_Float16)c4v;
#pragma unroll
            for (int i = 5; i < 16; ++i) { B1a.e[i] = (_Float16)0.0f; B1b.e[i] = (_Float16)0.0f; }
        }

        F8 c1a, c1b;
        c1a.v = __builtin_amdgcn_wmma_f32_16x16x32_f16(false, A1.v, false, B1a.v,
                                                       (short)0, zacc, false, false);
        c1b.v = __builtin_amdgcn_wmma_f32_16x16x32_f16(false, A1.v, false, B1b.v,
                                                       (short)0, zacc, false, false);
        float sa[8], sb[8];
#pragma unroll
        for (int j = 0; j < 8; ++j) {
            sa[j] = silu_f(c1a.e[j] + bb1[j]);
            sb[j] = silu_f(c1b.e[j] + bb1[j]);
        }
        H16 B2a, B2b;                      // own channels go into own K slots
#pragma unroll
        for (int j = 0; j < 8; ++j) {
            B2a.e[j] = (_Float16)sa[j];
            B2b.e[j] = (_Float16)sb[j];
        }
#pragma unroll
        for (int j = 8; j < 16; ++j) { B2a.e[j] = (_Float16)0.0f; B2b.e[j] = (_Float16)0.0f; }

        F8 c2a, c2b;
        c2a.v = __builtin_amdgcn_wmma_f32_16x16x32_f16(false, A2.v, false, B2a.v,
                                                       (short)0, zacc, false, false);
        c2b.v = __builtin_amdgcn_wmma_f32_16x16x32_f16(false, A2.v, false, B2b.v,
                                                       (short)0, zacc, false, false);
        float psa = 0.0f, psb = 0.0f;
#pragma unroll
        for (int j = 0; j < 8; ++j) {
            psa += lwv[j] * silu_f(c2a.e[j] + bb2[j]);
            psb += lwv[j] * silu_f(c2b.e[j] + bb2[j]);
        }
        psa += __shfl_xor(psa, 16);
        psb += __shfl_xor(psb, 16);
        if (!hi) {
            scoreL[(tlA + 0) * 256 + k] = msk ? -1e30f : (psa + lb0);
            scoreL[(tlA + 1) * 256 + k] = msk ? -1e30f : (psb + lb0);
        }
    }
    __syncthreads();

    // ---- phase 3: softmax over k per t-row (2 rows / wave) ----
    for (int rr = 0; rr < 2; ++rr) {
        const int tl = w * 2 + rr;
        const int t = t0 + tl;
        float vals[8];
        float mx = -1e30f;
#pragma unroll
        for (int i = 0; i < 8; ++i) {
            vals[i] = scoreL[tl * 256 + lane + 32 * i];
            mx = fmaxf(mx, vals[i]);
        }
        for (int o = 16; o > 0; o >>= 1) mx = fmaxf(mx, __shfl_xor(mx, o));
        float sum = 0.0f;
#pragma unroll
        for (int i = 0; i < 8; ++i) { vals[i] = __expf(vals[i] - mx); sum += vals[i]; }
        for (int o = 16; o > 0; o >>= 1) sum += __shfl_xor(sum, o);
        const float inv = fast_rcp(sum);
        const bool dead = (t >= mel_len) || (t >= T_);
#pragma unroll
        for (int i = 0; i < 8; ++i) {
            int k = lane + 32 * i;
            float wv = dead ? 0.0f : vals[i] * inv;
            scoreL[tl * 256 + k] = wv;
            WH[tl * 256 + k] = (_Float16)wv;
            if (t < T_) out_W[((size_t)(b * T_ + t)) * K_ + k] = wv;
        }
    }
    __syncthreads();

    // ---- phase 4: C path + ctx[t][2] = sum_k W*C (deterministic reduction) ----
    {
        const int tl = tid >> 4, kc4 = tid & 15;
        const int t = t0 + tl;
        const bool dead = (t >= mel_len) || (t >= T_);
        float a0 = 0.0f, a1 = 0.0f;
        if (!dead) {
            const float tv = (float)(t + 1);
            for (int p = 0; p < 16; ++p) {
                const int k = kc4 * 16 + p;
                const bool m = (maskL[k] != 0);
                const float S = m ? 0.0f : tv - skL[k];
                const float E = m ? 0.0f : ekL[k] - tv;
                F4 vc0, vc1;
                vc0.v = *(const v4f*)&bufA[2048 + k * 8];
                vc1.v = *(const v4f*)&bufA[2048 + k * 8 + 4];
                float h0 = cbc1[0] + S * cwc1[0] + E * cwc1[2];
                float h1v = cbc1[1] + S * cwc1[1] + E * cwc1[3];
#pragma unroll
                for (int c = 0; c < 4; ++c) {
                    h0  = fmaf(vc0.e[c], cwc1[(2 + c) * 2 + 0], h0);
                    h1v = fmaf(vc0.e[c], cwc1[(2 + c) * 2 + 1], h1v);
                    h0  = fmaf(vc1.e[c], cwc1[(6 + c) * 2 + 0], h0);
                    h1v = fmaf(vc1.e[c], cwc1[(6 + c) * 2 + 1], h1v);
                }
                h0 = silu_f(h0); h1v = silu_f(h1v);
                const float g0 = silu_f(cbc2[0] + h0 * cwc2[0] + h1v * cwc2[2]);
                const float g1 = silu_f(cbc2[1] + h0 * cwc2[1] + h1v * cwc2[3]);
                const float wv = scoreL[tl * 256 + k];
                a0 = fmaf(wv, g0, a0);
                a1 = fmaf(wv, g1, a1);
            }
        }
        cpart[(tl * 16 + kc4) * 2 + 0] = a0;
        cpart[(tl * 16 + kc4) * 2 + 1] = a1;
    }
    __syncthreads();
    if (tid < 32) {
        const int tl = tid >> 1, p = tid & 1;
        float s = 0.0f;
        for (int q = 0; q < 16; ++q) s += cpart[(tl * 16 + q) * 2 + p];
        ctx[tl * 2 + p] = s;
    }
    __syncthreads();

    // ---- phase 5: up = W @ V via WMMA; shared A fragment, two interleaved
    //      accumulator chains (d-tiles w and w+8) ----
    {
        const int d0 = w * 16 + lr;
        const int d1 = d0 + 128;
        F8 acc0, acc1; acc0.v = zacc; acc1.v = zacc;
        const _Float16* vtb0 = Vt_g + ((size_t)(b * D_ + d0)) * K_ + (hi ? 16 : 0);
        const _Float16* vtb1 = vtb0 + (size_t)128 * K_;
        const _Float16* whb  = &WH[lr * 256 + (hi ? 16 : 0)];
        for (int k0 = 0; k0 < K_; k0 += 32) {
            __builtin_prefetch(vtb0 + k0 + 32, 0, 1);
            __builtin_prefetch(vtb1 + k0 + 32, 0, 1);
            H8 alo, ahi;
            alo.v = *(const v8h*)(whb + k0);        // K = k0+(hi?16:0) .. +7
            ahi.v = *(const v8h*)(whb + k0 + 8);    // K = k0+8+(hi?16:0) .. +7
            H16 A;
#pragma unroll
            for (int i = 0; i < 8; ++i) { A.e[i] = alo.e[i]; A.e[8 + i] = ahi.e[i]; }
            H8 b0l, b0h, b1l, b1h;
            const v8h* p0 = reinterpret_cast<const v8h*>(vtb0 + k0);
            const v8h* p1 = reinterpret_cast<const v8h*>(vtb1 + k0);
            b0l.v = p0[0]; b0h.v = p0[1];
            b1l.v = p1[0]; b1h.v = p1[1];
            H16 Bm0, Bm1;
#pragma unroll
            for (int i = 0; i < 8; ++i) {
                Bm0.e[i] = b0l.e[i]; Bm0.e[8 + i] = b0h.e[i];
                Bm1.e[i] = b1l.e[i]; Bm1.e[8 + i] = b1h.e[i];
            }
            acc0.v = __builtin_amdgcn_wmma_f32_16x16x32_f16(false, A.v, false, Bm0.v,
                                                            (short)0, acc0.v, false, false);
            acc1.v = __builtin_amdgcn_wmma_f32_16x16x32_f16(false, A.v, false, Bm1.v,
                                                            (short)0, acc1.v, false, false);
        }
        const float le00 = le_g[d0], le10 = le_g[256 + d0], leb0 = leb_g[d0];
        const float le01 = le_g[d1], le11 = le_g[256 + d1], leb1 = leb_g[d1];
#pragma unroll
        for (int j = 0; j < 8; ++j) {
            const int tr = (hi ? 8 : 0) + j;
            const float c0 = ctx[tr * 2 + 0], c1 = ctx[tr * 2 + 1];
            bufA[tr * 256 + d0] = acc0.e[j] + c0 * le00 + c1 * le10 + leb0;
            bufA[tr * 256 + d1] = acc1.e[j] + c0 * le01 + c1 * le11 + leb1;
        }
    }
    __syncthreads();

    // ---- phase 6: LayerNorm over D + mel mask + store ----
    for (int rr = 0; rr < 2; ++rr) {
        const int tl = w * 2 + rr;
        const int t = t0 + tl;
        float x[8];
        float sum = 0.0f;
#pragma unroll
        for (int i = 0; i < 8; ++i) { x[i] = bufA[tl * 256 + lane + 32 * i]; sum += x[i]; }
        for (int o = 16; o > 0; o >>= 1) sum += __shfl_xor(sum, o);
        const float mu = sum * (1.0f / 256.0f);
        float vs = 0.0f;
#pragma unroll
        for (int i = 0; i < 8; ++i) { float dd = x[i] - mu; vs += dd * dd; }
        for (int o = 16; o > 0; o >>= 1) vs += __shfl_xor(vs, o);
        const float inv = fast_rsq(vs * (1.0f / 256.0f) + 1e-5f);
        const bool dead = (t >= mel_len);
        if (t < T_) {
            float* op = out_up + ((size_t)(b * T_ + t)) * D_;
#pragma unroll
            for (int i = 0; i < 8; ++i) {
                const int dd = lane + 32 * i;
                op[dd] = dead ? 0.0f : (x[i] - mu) * inv * lns_g[dd] + lnb_g[dd];
            }
        }
    }
}

// ---------------------------------------------------------------------------
extern "C" void kernel_launch(void* const* d_in, const int* in_sizes, int n_in,
                              void* d_out, int out_size, void* d_ws, size_t ws_size,
                              hipStream_t stream)
{
    const float* duration     = (const float*)d_in[0];
    const float* V            = (const float*)d_in[1];
    const int*   src_len      = (const int*)  d_in[2];
    // d_in[3] src_mask (recomputed from src_len), d_in[4] max_src_len unused
    const float* conv_w_kern  = (const float*)d_in[5];
    const float* conv_w_bias  = (const float*)d_in[6];
    const float* ln_w_scale   = (const float*)d_in[7];
    const float* ln_w_bias    = (const float*)d_in[8];
    const float* sw_w_w1      = (const float*)d_in[9];
    const float* sw_w_b1      = (const float*)d_in[10];
    const float* sw_w_w2      = (const float*)d_in[11];
    const float* sw_w_b2      = (const float*)d_in[12];
    const float* lin_w_w      = (const float*)d_in[13];
    const float* lin_w_b      = (const float*)d_in[14];
    const float* conv_c_kern  = (const float*)d_in[15];
    const float* conv_c_bias  = (const float*)d_in[16];
    const float* ln_c_scale   = (const float*)d_in[17];
    const float* ln_c_bias    = (const float*)d_in[18];
    const float* sw_c_w1      = (const float*)d_in[19];
    const float* sw_c_b1      = (const float*)d_in[20];
    const float* sw_c_w2      = (const float*)d_in[21];
    const float* sw_c_b2      = (const float*)d_in[22];
    const float* lin_e_w      = (const float*)d_in[23];
    const float* lin_e_b      = (const float*)d_in[24];
    const float* ln_out_scale = (const float*)d_in[25];
    const float* ln_out_bias  = (const float*)d_in[26];

    // workspace carve-up
    int*      mlen = (int*)d_ws;                       // 8 ints
    float*    sk   = (float*)d_ws + 16;                // 2048
    float*    ek   = sk + B_ * K_;                     // 2048
    float*    Vw   = ek + B_ * K_;                     // 16384
    float*    Vc   = Vw + B_ * K_ * 8;                 // 16384
    _Float16* Vt   = (_Float16*)(Vc + B_ * K_ * 8);    // 524288 halves

    // output carve-up (tuple: up, mel_mask, mel_len, W — flat, in order)
    float* out      = (float*)d_out;
    float* out_up   = out;
    float* out_mask = out + (size_t)B_ * T_ * D_;
    float* out_mlen = out_mask + (size_t)B_ * T_;
    float* out_W    = out_mlen + B_;

    k_prep<<<B_, 256, 0, stream>>>(duration, sk, ek, mlen, out_mask, out_mlen);
    k_conv<<<B_, 256, 0, stream>>>(V, conv_w_kern, conv_w_bias, ln_w_scale, ln_w_bias,
                                   conv_c_kern, conv_c_bias, ln_c_scale, ln_c_bias,
                                   Vw, Vc);
    k_tr<<<(B_ * K_ * D_) / 256, 256, 0, stream>>>(V, Vt);

    dim3 grid(NTILE, B_);
    k_main<<<grid, 256, 0, stream>>>(sk, ek, mlen, src_len, Vw, Vc, Vt,
                                     sw_w_w1, sw_w_b1, sw_w_w2, sw_w_b2,
                                     lin_w_w, lin_w_b,
                                     sw_c_w1, sw_c_b1, sw_c_w2, sw_c_b2,
                                     lin_e_w, lin_e_b, ln_out_scale, ln_out_bias,
                                     out_up, out_W);
}
